// MultiHeadAttentionDecoder_19061064859801
// MI455X (gfx1250) — compile-verified, matrix operand order checked by reading, not run on hardware
//
#include <hip/hip_runtime.h>

#define XD   64
#define ZD   128
#define HN   512
#define TAUK 16
#define DT   192      // XD + ZD
#define BB   32
#define TT   2048
#define KTOT 1152     // TAUK*XD + ZD
#define NKC  36       // KTOT / 32

#define XS_ROWS   272          // 256 t + 16 history
#define XS_STRIDE 72           // padded (bank-conflict-free, 16B aligned)
#define ZS_ROWS   256
#define ZS_STRIDE 136
#define GEMM_LDS_BYTES ((XS_ROWS*XS_STRIDE + ZS_ROWS*ZS_STRIDE) * 2)   // 108800

typedef __attribute__((ext_vector_type(16))) _Float16 v16h;
typedef __attribute__((ext_vector_type(8)))  float    v8f;
typedef __attribute__((ext_vector_type(4)))  float    v4f;

// ---------------------------------------------------------------------------
// Kernel 1: fused W1 transpose-copy + v/c precompute.
// One block per (i,k). Reads W1 once (402 MB), writes influence (402 MB,
// nontemporal), and accumulates v[i,k,d] = <W1[i,k,d,:], W2[i,k,:]> and
// c[i,k] = <b1[i,k,:], W2[i,k,:]> + b2[i,k].
// ---------------------------------------------------------------------------
__global__ void prep_kernel(const float* __restrict__ W1, const float* __restrict__ b1,
                            const float* __restrict__ W2, const float* __restrict__ b2,
                            float* __restrict__ infl, float* __restrict__ v_out,
                            float* __restrict__ c_out) {
  __shared__ float sW2[HN];
  __shared__ float v_lds[DT];
  __shared__ float c_lds;
  const int blk = blockIdx.x;            // blk = i*16 + k
  const int i = blk >> 4, k = blk & 15;
  const int tid = threadIdx.x;
  const int lane = tid & 31;

  sW2[tid]       = W2[(size_t)blk * HN + tid];
  sW2[tid + 256] = W2[(size_t)blk * HN + tid + 256];
  if (tid < DT) v_lds[tid] = 0.f;
  if (tid == 0) c_lds = 0.f;
  __syncthreads();

  const v4f* in4  = reinterpret_cast<const v4f*>(W1 + (size_t)blk * (DT * HN));
  v4f*       out4 = reinterpret_cast<v4f*>(infl + (size_t)(k * 64 + i) * (DT * HN));

  for (int p = 0; p < 96; ++p) {
    const int e4 = p * 256 + tid;        // float4 index within this (i,k) block
    v4f w = in4[e4];
    __builtin_nontemporal_store(w, out4 + e4);
    const int e = e4 * 4;
    const int d = e >> 9;                // 512 h per d; 4 elems never cross a d
    const int h = e & 511;
    float dot = w.x * sW2[h] + w.y * sW2[h + 1] + w.z * sW2[h + 2] + w.w * sW2[h + 3];
    #pragma unroll
    for (int off = 16; off > 0; off >>= 1) dot += __shfl_xor(dot, off, 32);
    if (lane == 0) atomicAdd(&v_lds[d], dot);   // whole wave shares one d
  }
  {
    const int h0 = tid * 2;
    float cp = b1[(size_t)blk * HN + h0] * sW2[h0] +
               b1[(size_t)blk * HN + h0 + 1] * sW2[h0 + 1];
    #pragma unroll
    for (int off = 16; off > 0; off >>= 1) cp += __shfl_xor(cp, off, 32);
    if (lane == 0) atomicAdd(&c_lds, cp);
  }
  __syncthreads();
  if (tid < DT) v_out[(size_t)blk * DT + tid] = v_lds[tid];
  if (tid == 0) c_out[blk] = c_lds + b2[blk];
}

// ---------------------------------------------------------------------------
// Kernel 2: pack the 1152x64 GEMM weight into WMMA B-fragment order (f16)
// and compute Csum[i] = sum_k c[i,k].
// B fragment layout (ISA 7.12.2, 16-bit B 32x16): lane<16 -> N=lane, K=0..7,16..23
// packed 2/VGPR; lane>=16 -> N=lane-16, K=8..15,24..31.
// Rows K<1024: Wg[K=k*64+d][i] = v[i,k,d]  (vx)
// Rows K>=1024: Wg[1024+d][i] = sum_k v[i,k,64+d]  (vz summed over k)
// ---------------------------------------------------------------------------
__global__ void pack_kernel(const float* __restrict__ v, const float* __restrict__ c,
                            _Float16* __restrict__ Wgf, float* __restrict__ Csum) {
  const int id = blockIdx.x * 256 + threadIdx.x;
  if (id < NKC * 4 * 32) {
    const int kc    = id / 128;
    const int rem   = id % 128;
    const int it    = rem / 32;
    const int lane  = rem % 32;
    const int i     = it * 16 + (lane & 15);
    const int khalf = lane >> 4;
    union { v16h h; unsigned u[8]; } frag;
    #pragma unroll
    for (int j = 0; j < 8; ++j) {
      const int Kb = kc * 32 + ((j < 4) ? 2 * j : 2 * j + 8) + khalf * 8;
      #pragma unroll
      for (int s = 0; s < 2; ++s) {
        const int K = Kb + s;
        float val;
        if (K < TAUK * XD) {
          const int k = K >> 6, d = K & 63;
          val = v[(size_t)(i * 16 + k) * DT + d];
        } else {
          const int dz = K - TAUK * XD;
          val = 0.f;
          #pragma unroll
          for (int k = 0; k < TAUK; ++k) val += v[(size_t)(i * 16 + k) * DT + XD + dz];
        }
        frag.h[j * 2 + s] = (_Float16)val;
      }
    }
    reinterpret_cast<v16h*>(Wgf)[id] = frag.h;
  } else if (id < NKC * 4 * 32 + 64) {
    const int i = id - NKC * 4 * 32;
    float s = 0.f;
    #pragma unroll
    for (int k = 0; k < TAUK; ++k) s += c[i * 16 + k];
    Csum[i] = s;
  }
}

// ---------------------------------------------------------------------------
// Kernel 3: WMMA GEMM. Block = 8 waves, block covers 256 consecutive t of one
// batch. Each wave owns a 32(t) x 64(i) tile (two 16-row M-tiles sharing B
// fragments): per K-chunk, 4 B loads feed 8 WMMAs, so one loadcnt wait is
// amortized over 8 matrix ops and B re-fetch traffic is halved.
// x window (272 rows) and z (256 rows) staged in dynamic LDS as f16 with
// padded strides (72 / 136 halves -> conflict-free 16B ds_load_b128 runs).
// Note: each 32-wide K-chunk lies in a single k band (k = kc>>1).
// ---------------------------------------------------------------------------
__global__ void gemm_kernel(const float* __restrict__ z, const float* __restrict__ x,
                            const _Float16* __restrict__ Wgf, const float* __restrict__ Csum,
                            float* __restrict__ out) {
  extern __shared__ _Float16 smem[];
  _Float16* xs = smem;                          // XS_ROWS * XS_STRIDE
  _Float16* zs = smem + XS_ROWS * XS_STRIDE;    // ZS_ROWS * ZS_STRIDE
  const int b    = blockIdx.x >> 3;
  const int tseg = (blockIdx.x & 7) * 256;
  const int tid  = threadIdx.x;

  for (int idx = tid; idx < XS_ROWS * XD; idx += 256) {
    const int row = idx >> 6, d = idx & 63;
    const int tg = tseg - 16 + row;
    const float val = (tg >= 0) ? x[((size_t)b * TT + tg) * XD + d] : 0.f;
    xs[row * XS_STRIDE + d] = (_Float16)val;
  }
  for (int idx = tid; idx < ZS_ROWS * ZD; idx += 256) {
    const int row = idx >> 7, d = idx & 127;
    zs[row * ZS_STRIDE + d] = (_Float16)z[((size_t)b * TT + tseg + row) * ZD + d];
  }
  __syncthreads();

  const int wave = tid >> 5, lane = tid & 31;
  const int m = lane & 15, khalf = lane >> 4;
  v8f acc[2][4] = {};
  const v16h* Bf = reinterpret_cast<const v16h*>(Wgf);

  for (int kc = 0; kc < NKC; ++kc) {
    v16h bf[4];
    #pragma unroll
    for (int it = 0; it < 4; ++it) bf[it] = Bf[(kc * 4 + it) * 32 + lane];

    union { v16h h; unsigned u[8]; } a[2];
    if (kc < 32) {                       // x-window chunks: A[t][K]=x[t-1-k][d]
      const int k = kc >> 1;             // entire chunk shares one k
      const int dbase = (kc & 1) * 32;
      #pragma unroll
      for (int mt = 0; mt < 2; ++mt) {
        const int row = wave * 32 + mt * 16 + m + 15 - k;   // LDS row of t-1-k
        #pragma unroll
        for (int j = 0; j < 8; ++j) {
          const int d = dbase + ((j < 4) ? 2 * j : 2 * j + 8) + khalf * 8;
          a[mt].u[j] = *reinterpret_cast<const unsigned*>(&xs[row * XS_STRIDE + d]);
        }
      }
    } else {                             // z chunks: A[t][1024+dz] = z[t][dz]
      const int dzbase = (kc - 32) * 32;
      #pragma unroll
      for (int mt = 0; mt < 2; ++mt) {
        const int row = wave * 32 + mt * 16 + m;
        #pragma unroll
        for (int j = 0; j < 8; ++j) {
          const int dz = dzbase + ((j < 4) ? 2 * j : 2 * j + 8) + khalf * 8;
          a[mt].u[j] = *reinterpret_cast<const unsigned*>(&zs[row * ZS_STRIDE + dz]);
        }
      }
    }
    #pragma unroll
    for (int mt = 0; mt < 2; ++mt)
      #pragma unroll
      for (int it = 0; it < 4; ++it)
        acc[mt][it] = __builtin_amdgcn_wmma_f32_16x16x32_f16(
            false, a[mt].h, false, bf[it], (short)0, acc[mt][it], false, false);
  }

  // C/D layout: lane<16 -> N=lane, M=r; lane>=16 -> N=lane-16, M=8+r
  #pragma unroll
  for (int it = 0; it < 4; ++it) {
    const int i = it * 16 + m;
    const float cs = Csum[i];
    #pragma unroll
    for (int mt = 0; mt < 2; ++mt) {
      #pragma unroll
      for (int r = 0; r < 8; ++r) {
        const int t = tseg + wave * 32 + mt * 16 + r + khalf * 8;
        out[((size_t)b * TT + t) * XD + i] = acc[mt][it][r] + cs;
      }
    }
  }
}

// ---------------------------------------------------------------------------
// Kernel 4: exact fp32 rewrite of the t<16 rows (partial causal validity):
// out[b,t,i] = sum_{k<t} ( x[b,t-1-k]·vx[i,k] + z[b,t]·vz[i,k] + c[i,k] )
// 32*16*64 = 32768 outputs; trivial cost, overwrites the GEMM's rows.
// ---------------------------------------------------------------------------
__global__ void fixup_kernel(const float* __restrict__ z, const float* __restrict__ x,
                             const float* __restrict__ v, const float* __restrict__ c,
                             float* __restrict__ out) {
  const int id = blockIdx.x * 256 + threadIdx.x;    // 32768 total
  const int i = id & 63;
  const int t = (id >> 6) & 15;
  const int b = id >> 10;
  float acc = 0.f;
  const float* zp = z + ((size_t)b * TT + t) * ZD;
  for (int k = 0; k < t; ++k) {
    const float* vp = v + (size_t)(i * 16 + k) * DT;
    const float* xp = x + ((size_t)b * TT + (t - 1 - k)) * XD;
    float s = c[i * 16 + k];
    for (int d = 0; d < XD; ++d) s += xp[d] * vp[d];
    for (int d = 0; d < ZD; ++d) s += zp[d] * vp[XD + d];
    acc += s;
  }
  out[((size_t)b * TT + t) * XD + i] = acc;
}

// ---------------------------------------------------------------------------
extern "C" void kernel_launch(void* const* d_in, const int* in_sizes, int n_in,
                              void* d_out, int out_size, void* d_ws, size_t ws_size,
                              hipStream_t stream) {
  const float* z  = (const float*)d_in[0];
  const float* x  = (const float*)d_in[1];
  const float* W1 = (const float*)d_in[2];
  const float* b1 = (const float*)d_in[3];
  const float* W2 = (const float*)d_in[4];
  const float* b2 = (const float*)d_in[5];

  float* out  = (float*)d_out;                       // (32,2048,64)
  float* infl = out + (size_t)BB * TT * XD;          // (16,64,192,512)

  float*    ws   = (float*)d_ws;
  float*    v    = ws;                               // 64*16*192 = 196608 f32
  float*    c    = ws + 196608;                      // 1024 f32
  float*    Csum = ws + 197632;                      // 64 f32
  _Float16* Wgf  = (_Float16*)(ws + 197696);         // 36*4*32*16 = 73728 f16 (32B-aligned)

  // Allow >64KB dynamic LDS for the GEMM (CDNA5: up to 320KB per workgroup).
  (void)hipFuncSetAttribute(reinterpret_cast<const void*>(gemm_kernel),
                            hipFuncAttributeMaxDynamicSharedMemorySize,
                            GEMM_LDS_BYTES);

  prep_kernel <<<1024, 256, 0, stream>>>(W1, b1, W2, b2, infl, v, c);
  pack_kernel <<<19,   256, 0, stream>>>(v, c, Wgf, Csum);
  gemm_kernel <<<256,  256, GEMM_LDS_BYTES, stream>>>(z, x, Wgf, Csum, out);
  fixup_kernel<<<128,  256, 0, stream>>>(z, x, v, c, out);
}
